// GAT_87668872446714
// MI455X (gfx1250) — matrix operand order repaired and checked by reading, not compiled, not run on hardware
//
#include <hip/hip_runtime.h>
#include <hip/hip_bf16.h>

// ---------------------------------------------------------------------------
// GAT (2-layer, dense adjacency) for gfx1250 using v_wmma_f32_16x16x32_f16.
// N=4096, F_IN=512, F_HID=256, F_CLS=16, H=4, alpha=0.2
// ---------------------------------------------------------------------------

typedef __attribute__((ext_vector_type(16))) _Float16 v16h;
typedef __attribute__((ext_vector_type(8)))  _Float16 v8h;
typedef __attribute__((ext_vector_type(8)))  float    v8f;

#define NEG_BIG (-9.0e15f)

static __device__ __forceinline__ v16h cat8(v8h lo, v8h hi) {
  v16h r;
#pragma unroll
  for (int i = 0; i < 8; ++i) { r[i] = lo[i]; r[i + 8] = hi[i]; }
  return r;
}

// ------------------------- conversion kernels ------------------------------

__global__ void k_f32_to_f16(const float* __restrict__ x, _Float16* __restrict__ y, long n) {
  long i = (long)blockIdx.x * blockDim.x + threadIdx.x;
  long stride = (long)gridDim.x * blockDim.x;
  for (; i < n; i += stride) y[i] = (_Float16)x[i];
}

// src[b][R][C] f32 -> dst[b][C][R] f16
__global__ void k_transpose_to_f16(const float* __restrict__ src, _Float16* __restrict__ dst,
                                   int R, int C) {
  int b = blockIdx.y;
  long i = (long)blockIdx.x * blockDim.x + threadIdx.x;
  long total = (long)R * C;
  if (i >= total) return;
  int rr = (int)(i / C), cc = (int)(i % C);
  dst[(long)b * total + (long)cc * R + rr] = (_Float16)src[(long)b * total + i];
}

// ------------------------- WMMA GEMM ---------------------------------------
// C_T = (A * B^T)^T stored as f16, with A [M][K] f16 row-major (shared over
// batch), BT [batch][Ncols][K] f16 (feature-major), output CT [batch][Ncols][M].
// One wave computes a 16 x (CTILES*16) tile.
template <int CTILES>
__global__ void k_gemm_f16_wmma(const _Float16* __restrict__ A,
                                const _Float16* __restrict__ BT,
                                _Float16* __restrict__ CT,
                                int M, int K, int Ncols) {
  const int b   = blockIdx.z;
  const int i0  = blockIdx.x * 16;
  const int c0  = blockIdx.y * (CTILES * 16);
  const int ln  = threadIdx.x;
  const int r   = ln & 15;
  const int hi  = ln >> 4;

  const _Float16* Ar = A + (long)(i0 + r) * K;
  const _Float16* Bb = BT + (long)b * Ncols * K;

  v8f acc[CTILES];
#pragma unroll
  for (int t = 0; t < CTILES; ++t)
#pragma unroll
    for (int v = 0; v < 8; ++v) acc[t][v] = 0.0f;

  for (int k0 = 0; k0 < K; k0 += 32) {
    // A layout: lanes 0-15 hold K = k0+0..7 & k0+16..23; lanes 16-31 K+8.
    v8h a0 = *(const v8h*)(Ar + k0 + hi * 8);
    v8h a1 = *(const v8h*)(Ar + k0 + 16 + hi * 8);
    v16h a = cat8(a0, a1);
#pragma unroll
    for (int t = 0; t < CTILES; ++t) {
      const _Float16* bp = Bb + (long)(c0 + t * 16 + r) * K + k0;
      v16h bb = cat8(*(const v8h*)(bp + hi * 8), *(const v8h*)(bp + 16 + hi * 8));
      acc[t] = __builtin_amdgcn_wmma_f32_16x16x32_f16(
          false, a, false, bb, (short)0, acc[t], false, false);
    }
  }

  _Float16* Cb = CT + (long)b * Ncols * M;
#pragma unroll
  for (int t = 0; t < CTILES; ++t)
#pragma unroll
    for (int v = 0; v < 8; ++v)
      Cb[(long)(c0 + t * 16 + r) * M + i0 + v + 8 * hi] = (_Float16)acc[t][v];
}

// ------------------------- edge scores -------------------------------------
// e_src[h][n] = sum_o hT[h][o][n]*a[h][o]; e_dst uses a[h][O+o].
__global__ void k_edge_scores(const _Float16* __restrict__ hT, const float* __restrict__ a,
                              float* __restrict__ es, float* __restrict__ ed,
                              int N, int O) {
  int n = blockIdx.x * blockDim.x + threadIdx.x;
  int h = blockIdx.y;
  if (n >= N) return;
  const _Float16* p = hT + (long)h * O * N + n;
  const float* ah = a + (long)h * 2 * O;
  float s = 0.f, d = 0.f;
  for (int o = 0; o < O; ++o) {
    float v = (float)p[(long)o * N];
    s += v * ah[o];
    d += v * ah[O + o];
  }
  es[(long)h * N + n] = s;
  ed[(long)h * N + n] = d;
}

// ------------------------- flash attention + aggregation -------------------
// One wave handles 16 rows x (NT*16) output cols of head blockIdx.y.
// Online softmax over j with masked leaky-relu logits; P*H via WMMA.
template <int NT>
__global__ void k_attn(const float* __restrict__ e_src, const float* __restrict__ e_dst,
                       const _Float16* __restrict__ hT,   // [H][NT*16][N]
                       const int* __restrict__ adj,       // [N][N]
                       float* __restrict__ out,           // [H][N][NT*16]
                       int N, float alpha) {
  const int h  = blockIdx.y;
  const int i0 = blockIdx.x * 16;
  const int ln = threadIdx.x;
  const int r  = ln & 15;
  const int hi = ln >> 4;
  const int row = i0 + r;

  const float es = e_src[(long)h * N + row];
  const float* edh = e_dst + (long)h * N;
  const int* adjr = adj + (long)row * N;
  const _Float16* hTh = hT + (long)h * (NT * 16) * N;

  float m = -INFINITY;
  float l = 0.0f;
  v8f acc[NT];
#pragma unroll
  for (int t = 0; t < NT; ++t)
#pragma unroll
    for (int v = 0; v < 8; ++v) acc[t][v] = 0.0f;

  for (int jb = 0; jb < N; jb += 32) {
    // logits in WMMA A layout: this lane covers K = hi*8+t and 16+hi*8+t
    float ev[16];
#pragma unroll
    for (int t = 0; t < 8; ++t) {
      int j0 = jb + hi * 8 + t;
      int j1 = j0 + 16;
      float v0 = es + edh[j0];
      float v1 = es + edh[j1];
      v0 = v0 > 0.f ? v0 : alpha * v0;
      v1 = v1 > 0.f ? v1 : alpha * v1;
      ev[t]     = adjr[j0] != 0 ? v0 : NEG_BIG;
      ev[t + 8] = adjr[j1] != 0 ? v1 : NEG_BIG;
    }
    float mloc = -INFINITY;
#pragma unroll
    for (int t = 0; t < 16; ++t) mloc = fmaxf(mloc, ev[t]);
    mloc = fmaxf(mloc, __shfl_xor(mloc, 16));       // combine the two K-halves

    float mnew = fmaxf(m, mloc);
    float scale = __expf(m - mnew);
    float ls = 0.f;
    v16h pa;
#pragma unroll
    for (int t = 0; t < 16; ++t) {
      float p = __expf(ev[t] - mnew);
      ls += p;
      pa[t] = (_Float16)p;
    }
    ls += __shfl_xor(ls, 16);
    l = l * scale + ls;
    m = mnew;

    // broadcast per-row rescale to C-layout rows (row = v + 8*hi)
    float rs[8];
#pragma unroll
    for (int v = 0; v < 8; ++v) rs[v] = __shfl(scale, v + 8 * hi);
#pragma unroll
    for (int t = 0; t < NT; ++t)
#pragma unroll
      for (int v = 0; v < 8; ++v) acc[t][v] *= rs[v];

    const int kbase = jb;                            // B layout: K = hi*8+t and 16+hi*8+t
#pragma unroll
    for (int t = 0; t < NT; ++t) {
      const _Float16* bp = hTh + (long)(t * 16 + r) * N + kbase;
      v16h bb = cat8(*(const v8h*)(bp + hi * 8), *(const v8h*)(bp + 16 + hi * 8));
      acc[t] = __builtin_amdgcn_wmma_f32_16x16x32_f16(
          false, pa, false, bb, (short)0, acc[t], false, false);
    }
  }

  float linv = 1.0f / l;
  float rl[8];
#pragma unroll
  for (int v = 0; v < 8; ++v) rl[v] = __shfl(linv, v + 8 * hi);

  const int O = NT * 16;
#pragma unroll
  for (int t = 0; t < NT; ++t)
#pragma unroll
    for (int v = 0; v < 8; ++v)
      out[((long)h * N + i0 + v + 8 * hi) * O + t * 16 + r] = acc[t][v] * rl[v];
}

// ------------------------- head mean + relu -> f16 -------------------------
__global__ void k_mean_relu_f16(const float* __restrict__ out1, _Float16* __restrict__ hmid,
                                int N, int O, int H) {
  long i = (long)blockIdx.x * blockDim.x + threadIdx.x;
  long total = (long)N * O;
  if (i >= total) return;
  float s = 0.f;
  for (int h = 0; h < H; ++h) s += out1[(long)h * total + i];
  s *= (1.0f / H);
  hmid[i] = (_Float16)(s > 0.f ? s : 0.f);
}

// ------------------------- log-softmax over classes ------------------------
__global__ void k_log_softmax(const float* __restrict__ in, float* __restrict__ out,
                              int N, int C) {
  int n = blockIdx.x * blockDim.x + threadIdx.x;
  if (n >= N) return;
  const float* p = in + (long)n * C;
  float m = -INFINITY;
  for (int c = 0; c < C; ++c) m = fmaxf(m, p[c]);
  float s = 0.f;
  for (int c = 0; c < C; ++c) s += __expf(p[c] - m);
  float lse = __logf(s) + m;
  for (int c = 0; c < C; ++c) out[(long)n * C + c] = p[c] - lse;
}

// ---------------------------------------------------------------------------

extern "C" void kernel_launch(void* const* d_in, const int* in_sizes, int n_in,
                              void* d_out, int out_size, void* d_ws, size_t ws_size,
                              hipStream_t stream) {
  (void)in_sizes; (void)n_in; (void)out_size; (void)ws_size;

  const int N = 4096, FIN = 512, FH = 256, FC = 16, H = 4;
  const float alpha = 0.2f;

  const float* x  = (const float*)d_in[0];
  const int*   adj = (const int*)d_in[1];
  const float* W1 = (const float*)d_in[2];
  const float* a1 = (const float*)d_in[3];
  const float* W2 = (const float*)d_in[4];
  const float* a2 = (const float*)d_in[5];
  float* outp = (float*)d_out;

  // carve workspace (256B aligned slices)
  size_t off = 0;
  auto carve = [&](size_t bytes) -> void* {
    void* p = (char*)d_ws + off;
    off += (bytes + 255) & ~(size_t)255;
    return p;
  };
  _Float16* xf16  = (_Float16*)carve((size_t)N * FIN * 2);          // 4 MB
  _Float16* wt1   = (_Float16*)carve((size_t)H * FIN * FH * 2);     // 1 MB  [h][o][f]
  _Float16* ht1   = (_Float16*)carve((size_t)H * FH * N * 2);       // 8 MB  [h][o][n]
  float*    e1s   = (float*)carve((size_t)H * N * 4);
  float*    e1d   = (float*)carve((size_t)H * N * 4);
  float*    out1  = (float*)carve((size_t)H * N * FH * 4);          // 16 MB
  _Float16* hmid  = (_Float16*)carve((size_t)N * FH * 2);           // 2 MB  [n][o]
  _Float16* wt2   = (_Float16*)carve((size_t)FH * FC * 2);          //       [o][f]
  _Float16* ht2   = (_Float16*)carve((size_t)FC * N * 2);           //       [o][n]
  float*    e2s   = (float*)carve((size_t)N * 4);
  float*    e2d   = (float*)carve((size_t)N * 4);
  float*    out2  = (float*)carve((size_t)N * FC * 4);

  // 1) precision conversions / transposes
  k_f32_to_f16<<<512, 256, 0, stream>>>(x, xf16, (long)N * FIN);
  k_transpose_to_f16<<<dim3((FIN * FH + 255) / 256, H), 256, 0, stream>>>(W1, wt1, FIN, FH);
  k_transpose_to_f16<<<dim3((FH * FC + 255) / 256, 1), 256, 0, stream>>>(W2, wt2, FH, FC);

  // 2) layer-1 feature GEMM: ht1[h][o][n] = (x @ W1[h])^T, f16
  k_gemm_f16_wmma<4><<<dim3(N / 16, FH / 64, H), 32, 0, stream>>>(xf16, wt1, ht1, N, FIN, FH);

  // 3) attention logits' row/col scores
  k_edge_scores<<<dim3(N / 256, H), 256, 0, stream>>>(ht1, a1, e1s, e1d, N, FH);

  // 4) masked softmax + aggregation (flash style), per head
  k_attn<16><<<dim3(N / 16, H), 32, 0, stream>>>(e1s, e1d, ht1, adj, out1, N, alpha);

  // 5) mean over heads + relu -> f16 row-major (layer-2 A operand)
  k_mean_relu_f16<<<(int)(((long)N * FH + 255) / 256), 256, 0, stream>>>(out1, hmid, N, FH, H);

  // 6) layer-2 feature GEMM: ht2[o][n]
  k_gemm_f16_wmma<1><<<dim3(N / 16, 1, 1), 32, 0, stream>>>(hmid, wt2, ht2, N, FH, FC);

  // 7) layer-2 scores + attention
  k_edge_scores<<<dim3(N / 256, 1), 256, 0, stream>>>(ht2, a2, e2s, e2d, N, FC);
  k_attn<1><<<dim3(N / 16, 1), 32, 0, stream>>>(e2s, e2d, ht2, adj, out2, N, alpha);

  // 8) log-softmax over the 16 classes
  k_log_softmax<<<N / 256, 256, 0, stream>>>(out2, outp, N, FC);
}
